// TableAttentionHead_8186207666755
// MI455X (gfx1250) — matrix operand order repaired and checked by reading, not compiled, not run on hardware
//
#include <hip/hip_runtime.h>
#include <stdint.h>

// ---------------------------------------------------------------------------
// TableAttentionHead for MI455X (gfx1250, wave32, WMMA).
//   B=128, L=256, IN=512, H=256, T=501, ELEM=30
// Pipeline (all on `stream`):
//   1) convert fea -> f16 (row-major + transposed copy), weights -> f16 padded
//   2) WMMA GEMM: bHp = fea @ W_i2h^T            [32768 x 256] f16
//   3) persistent decode: 32 WGs x 4 batch rows, all 501 steps in-kernel
//      (weights amortized over 4 rows -> 4x less L2 traffic)
//   4) WMMA GEMM + softmax: structure_probs -> d_out[0 .. B*T*30)
//   5) WMMA GEMM (W_loc_trans @ fea[b]) fused with loc_gen dot + sigmoid,
//      A-tile staged via global_load_async_to_lds_b128 (ASYNCcnt path)
// ---------------------------------------------------------------------------

typedef __attribute__((ext_vector_type(16))) _Float16 v16h;
typedef __attribute__((ext_vector_type(8)))  _Float16 v8h;
typedef __attribute__((ext_vector_type(8)))  float    v8f;

#define B_SZ     128
#define L_SZ     256
#define IN_SZ    512
#define H_SZ     256
#define T_STEPS  501
#define ELEM_N   30
#define XK       544   // IN_SZ + ELEM_N padded to multiple of 32
#define R_PB     4     // batch rows per decode workgroup
#define LOG2E    1.4426950408889634f

// ---------------- fragment helpers (ISA 7.12.2, wave32) --------------------
// A-matrix 16x32 f16, this code's contiguous k order (not the ISA table order; B matches): lane = 16*g + m ; VGPR v holds K pair
//   k = 16*g + 2*v + p
__device__ __forceinline__ v16h load_a_frag(const _Float16* row, int kb, int g) {
  v16h a;
  uint32_t* ap = (uint32_t*)&a;
#pragma unroll
  for (int v = 0; v < 8; ++v) {
    int k0 = kb + (g << 4) + (v << 1);
    ap[v] = *(const uint32_t*)(row + k0);
  }
  return a;
}

// B-matrix 32x16 f16 stored [N][K] row-major: lane = 16*g + n holds
// K = kb + 16*g .. +15 contiguous.
__device__ __forceinline__ v16h load_b_frag_nk(const _Float16* wb, int n, int K,
                                               int kb, int g) {
  return *(const v16h*)(wb + (size_t)n * K + kb + (g << 4));
}

__device__ __forceinline__ v8f wmma_f16(v16h a, v16h b, v8f c) {
  return __builtin_amdgcn_wmma_f32_16x16x32_f16(false, a, false, b,
                                                (short)0, c, false, false);
}

__device__ __forceinline__ float fast_tanh(float v) {
  v = fminf(fmaxf(v, -15.f), 15.f);
  float e = __builtin_amdgcn_exp2f(v * (-2.f * LOG2E));   // exp(-2v)
  return (1.f - e) / (1.f + e);
}
__device__ __forceinline__ float fast_sigmoid(float v) {
  return 1.f / (1.f + __builtin_amdgcn_exp2f(-v * LOG2E));
}

// ---- CDNA5 async load to LDS (ASYNCcnt) -----------------------------------
__device__ __forceinline__ void async_ld_b128(uint32_t lds_off, const void* g) {
  asm volatile("global_load_async_to_lds_b128 %0, %1, off"
               :: "v"(lds_off), "v"(g) : "memory");
}
__device__ __forceinline__ void wait_asynccnt0() {
  asm volatile("s_wait_asynccnt 0x0" ::: "memory");
}

// ---------------- conversion / padding / transpose -------------------------
__global__ __launch_bounds__(256) void cvt_f32_f16(_Float16* dst, const float* src,
                                                   int n) {
  for (int i = blockIdx.x * 256 + threadIdx.x; i < n; i += gridDim.x * 256)
    dst[i] = (_Float16)src[i];
}

__global__ __launch_bounds__(256) void cvt_pad(_Float16* dst, const float* src,
                                               int drows, int dcols,
                                               int srows, int scols) {
  int i = blockIdx.x * 256 + threadIdx.x;
  if (i >= drows * dcols) return;
  int r = i / dcols, c = i - r * dcols;
  dst[i] = (r < srows && c < scols) ? (_Float16)src[r * scols + c]
                                    : (_Float16)0.f;
}

// feaT[b][c][l] = (f16) fea[b][l][c] ; 32x32 LDS tile transpose
__global__ __launch_bounds__(256) void transpose_fea(const float* fea,
                                                     _Float16* feaT) {
  __shared__ _Float16 tile[32][33];
  int bid = blockIdx.x;                 // B * (L/32) * (IN/32) = 128*8*16
  int b = bid >> 7, tl = (bid >> 4) & 7, tc = bid & 15;
  int r = threadIdx.x >> 5, c = threadIdx.x & 31;
  const float* src = fea + ((size_t)b * L_SZ + tl * 32) * IN_SZ + tc * 32;
#pragma unroll
  for (int p = 0; p < 4; ++p) {
    int rr = p * 8 + r;
    tile[rr][c] = (_Float16)src[(size_t)rr * IN_SZ + c];
  }
  __syncthreads();
  _Float16* dst = feaT + ((size_t)b * IN_SZ + tc * 32) * L_SZ + tl * 32;
#pragma unroll
  for (int p = 0; p < 4; ++p) {
    int rr = p * 8 + r;
    dst[(size_t)rr * L_SZ + c] = tile[c][rr];
  }
}

// ---------------- 2) batch_H_proj GEMM (WMMA) ------------------------------
// bHp[m][n] = sum_k fea_h[m][k] * Wi2h[n][k] ; M=32768 N=256 K=512
__global__ __launch_bounds__(256) void bhp_gemm(const _Float16* A,
                                                const _Float16* Wn_k,
                                                _Float16* D) {
  int wave = threadIdx.x >> 5, lane = threadIdx.x & 31;
  int tile = blockIdx.x * 8 + wave;          // 32768 tiles
  int mt = tile >> 4, nt = tile & 15;
  int g = lane >> 4, mn = lane & 15;
  const _Float16* arow = A + (size_t)(mt * 16 + mn) * IN_SZ;
  v8f c = {};
  for (int ks = 0; ks < 16; ++ks) {
    v16h a = load_a_frag(arow, ks * 32, g);
    v16h b = load_b_frag_nk(Wn_k, nt * 16 + mn, IN_SZ, ks * 32, g);
    c = wmma_f16(a, b, c);
  }
#pragma unroll
  for (int r = 0; r < 8; ++r) {
    int m = mt * 16 + r + (g << 3);
    D[(size_t)m * H_SZ + nt * 16 + mn] = (_Float16)c[r];
  }
}

// ---------------- 3) persistent sequential decode (4 rows / WG) ------------
__global__ __launch_bounds__(256) void decode_steps(
    const _Float16* fea_h, const _Float16* bHp,
    const _Float16* Wh2h,                       // [256][256] f16 (n,k)
    const float* b_h2h, const float* w_score,
    const _Float16* Wih,                        // [768][544] f16
    const _Float16* Whh,                        // [768][256] f16
    const float* b_ih, const float* b_hh,
    const _Float16* Wgen,                       // [32][256] f16
    const float* b_gen, _Float16* hs_h) {
  const int tid = threadIdx.x;
  const int b0 = blockIdx.x * R_PB;
  __shared__ __align__(16) float h[R_PB][H_SZ];
  __shared__ __align__(16) float q[R_PB][H_SZ];
  __shared__ float e[R_PB][L_SZ];                // scores, then alpha in-place
  __shared__ __align__(16) float xg[R_PB][XK];
  __shared__ float gi_s[R_PB][3 * H_SZ], gh_s[R_PB][3 * H_SZ];
  __shared__ float red[R_PB * 32], mxv[R_PB], inv[R_PB];
  __shared__ float lgt[R_PB][32];
  __shared__ float ws_s[H_SZ];
  __shared__ int elem[R_PB];

#pragma unroll
  for (int r = 0; r < R_PB; ++r) h[r][tid] = 0.f;
  ws_s[tid] = w_score[tid];
  if (tid < R_PB) elem[tid] = 0;
  __syncthreads();

  for (int t = 0; t < T_STEPS; ++t) {
    // ---- q[r] = h[r] @ Wh2h^T + b_h2h : weight row read once per thread
    {
      const _Float16* wr = Wh2h + (size_t)tid * H_SZ;
      float acc[R_PB];
#pragma unroll
      for (int r = 0; r < R_PB; ++r) acc[r] = b_h2h[tid];
      for (int k = 0; k < H_SZ; k += 8) {
        v8h w = *(const v8h*)(wr + k);
#pragma unroll
        for (int r = 0; r < R_PB; ++r) {
          const float* hr = &h[r][k];
#pragma unroll
          for (int j = 0; j < 8; ++j) acc[r] += hr[j] * (float)w[j];
        }
      }
#pragma unroll
      for (int r = 0; r < R_PB; ++r) q[r][tid] = acc[r];
    }
    __syncthreads();
    // ---- e[r][l] = w_score . tanh(bHp[b0+r, l, :] + q[r])  (thread = l)
#pragma unroll
    for (int r = 0; r < R_PB; ++r) {
      const _Float16* bl = bHp + ((size_t)(b0 + r) * L_SZ + tid) * H_SZ;
      float acc = 0.f;
      for (int hh = 0; hh < H_SZ; hh += 8) {
        v8h bv = *(const v8h*)(bl + hh);
#pragma unroll
        for (int j = 0; j < 8; ++j)
          acc += fast_tanh((float)bv[j] + q[r][hh + j]) * ws_s[hh + j];
      }
      e[r][tid] = acc;
    }
    __syncthreads();
    // ---- softmax over L per row
    if (tid < R_PB * 32) {
      int r = tid >> 5, j = tid & 31;
      float m = -1e30f;
#pragma unroll
      for (int i = 0; i < 8; ++i) m = fmaxf(m, e[r][j * 8 + i]);
      red[tid] = m;
    }
    __syncthreads();
    if (tid < R_PB) {
      float m = red[tid * 32];
      for (int i = 1; i < 32; ++i) m = fmaxf(m, red[tid * 32 + i]);
      mxv[tid] = m;
    }
    __syncthreads();
#pragma unroll
    for (int r = 0; r < R_PB; ++r)
      e[r][tid] = __builtin_amdgcn_exp2f((e[r][tid] - mxv[r]) * LOG2E);
    __syncthreads();
    if (tid < R_PB * 32) {
      int r = tid >> 5, j = tid & 31;
      float s = 0.f;
#pragma unroll
      for (int i = 0; i < 8; ++i) s += e[r][j * 8 + i];
      red[tid] = s;
    }
    __syncthreads();
    if (tid < R_PB) {
      float s = 0.f;
      for (int i = 0; i < 32; ++i) s += red[tid * 32 + i];
      inv[tid] = 1.f / s;
    }
    __syncthreads();
#pragma unroll
    for (int r = 0; r < R_PB; ++r) e[r][tid] *= inv[r];   // e := alpha
    __syncthreads();
    // ---- context: 4 groups of 64 threads, 8 cols each, coalesced per l
    {
      int r = tid >> 6, c0 = (tid & 63) * 8;
      const _Float16* feaR = fea_h + (size_t)(b0 + r) * L_SZ * IN_SZ;
      float acc[8] = {};
      for (int l = 0; l < L_SZ; ++l) {
        if (l + 8 < L_SZ)
          __builtin_prefetch(feaR + (size_t)(l + 8) * IN_SZ + c0, 0, 0);
        float al = e[r][l];
        v8h fv = *(const v8h*)(feaR + (size_t)l * IN_SZ + c0);
#pragma unroll
        for (int j = 0; j < 8; ++j) acc[j] += al * (float)fv[j];
      }
#pragma unroll
      for (int j = 0; j < 8; ++j) xg[r][c0 + j] = acc[j];
    }
    if (tid < R_PB * 32) {
      int r = tid >> 5, j = tid & 31;
      xg[r][IN_SZ + j] = (j == elem[r]) ? 1.f : 0.f;      // onehot + pad
    }
    __syncthreads();
    // ---- GRU gates: weight rows shared across the 4 batch rows
#pragma unroll
    for (int part = 0; part < 3; ++part) {
      int n = part * H_SZ + tid;
      float gi[R_PB], gh[R_PB];
#pragma unroll
      for (int r = 0; r < R_PB; ++r) { gi[r] = b_ih[n]; gh[r] = b_hh[n]; }
      const _Float16* wi = Wih + (size_t)n * XK;
      for (int k = 0; k < XK; k += 8) {
        v8h w = *(const v8h*)(wi + k);
#pragma unroll
        for (int r = 0; r < R_PB; ++r) {
          const float* xr = &xg[r][k];
#pragma unroll
          for (int j = 0; j < 8; ++j) gi[r] += xr[j] * (float)w[j];
        }
      }
      const _Float16* wh = Whh + (size_t)n * H_SZ;
      for (int k = 0; k < H_SZ; k += 8) {
        v8h w = *(const v8h*)(wh + k);
#pragma unroll
        for (int r = 0; r < R_PB; ++r) {
          const float* hr = &h[r][k];
#pragma unroll
          for (int j = 0; j < 8; ++j) gh[r] += hr[j] * (float)w[j];
        }
      }
#pragma unroll
      for (int r = 0; r < R_PB; ++r) { gi_s[r][n] = gi[r]; gh_s[r][n] = gh[r]; }
    }
    __syncthreads();
    // ---- hidden update
#pragma unroll
    for (int r = 0; r < R_PB; ++r) {
      float rg = fast_sigmoid(gi_s[r][tid] + gh_s[r][tid]);
      float z  = fast_sigmoid(gi_s[r][H_SZ + tid] + gh_s[r][H_SZ + tid]);
      float ng = fast_tanh(gi_s[r][2 * H_SZ + tid] + rg * gh_s[r][2 * H_SZ + tid]);
      float hn = (1.f - z) * ng + z * h[r][tid];
      h[r][tid] = hn;
      hs_h[((size_t)(b0 + r) * T_STEPS + t) * H_SZ + tid] = (_Float16)hn;
    }
    __syncthreads();
    // ---- logits + argmax per row
    if (tid < R_PB * 32) {
      int r = tid >> 5, j = tid & 31;
      if (j < ELEM_N) {
        float lg = b_gen[j];
        const _Float16* wr = Wgen + (size_t)j * H_SZ;
        for (int k = 0; k < H_SZ; k += 8) {
          v8h w = *(const v8h*)(wr + k);
#pragma unroll
          for (int u = 0; u < 8; ++u) lg += h[r][k + u] * (float)w[u];
        }
        lgt[r][j] = lg;
      }
    }
    __syncthreads();
    if (tid < R_PB) {
      int best = 0;
      float bv = lgt[tid][0];
      for (int j = 1; j < ELEM_N; ++j)
        if (lgt[tid][j] > bv) { bv = lgt[tid][j]; best = j; }
      elem[tid] = best;
    }
    __syncthreads();
  }
}

// ---------------- 4) structure_probs (WMMA + softmax) ----------------------
// M = B*T = 64128 (4008 tiles exactly), N = 32 (j<30 used), K = 256.
__global__ __launch_bounds__(256) void struct_probs_kernel(
    const _Float16* hs, const _Float16* Wgen, const float* b_gen, float* out) {
  __shared__ float St[4][16][32];
  int wave = threadIdx.x >> 5, lane = threadIdx.x & 31;
  int slot = wave >> 1, nt = wave & 1;
  int mt = blockIdx.x * 4 + slot;
  int g = lane >> 4, mn = lane & 15;
  const _Float16* arow = hs + (size_t)(mt * 16 + mn) * H_SZ;
  v8f c = {};
  for (int ks = 0; ks < 8; ++ks) {
    v16h a = load_a_frag(arow, ks * 32, g);
    v16h b = load_b_frag_nk(Wgen, nt * 16 + mn, H_SZ, ks * 32, g);
    c = wmma_f16(a, b, c);
  }
#pragma unroll
  for (int r = 0; r < 8; ++r) St[slot][r + (g << 3)][nt * 16 + mn] = c[r];
  __syncthreads();
  if (threadIdx.x < 64) {
    int s = threadIdx.x >> 4, r = threadIdx.x & 15;
    size_t m = (size_t)(blockIdx.x * 4 + s) * 16 + r;
    float lg[ELEM_N], mx = -1e30f;
    for (int j = 0; j < ELEM_N; ++j) {
      lg[j] = St[s][r][j] + b_gen[j];
      mx = fmaxf(mx, lg[j]);
    }
    float sum = 0.f;
    for (int j = 0; j < ELEM_N; ++j) {
      lg[j] = __builtin_amdgcn_exp2f((lg[j] - mx) * LOG2E);
      sum += lg[j];
    }
    float invs = 1.f / sum;
    for (int j = 0; j < ELEM_N; ++j) out[m * ELEM_N + j] = lg[j] * invs;
  }
}

// ---------------- 5) loc branch: WMMA GEMM fused with loc_gen --------------
// Dt[16][512] = W_lt[mt*16..+16, :] @ fea[b] (WMMA, A staged async to LDS),
// out2[b,t,j] = sigmoid(hs.W2[:256] + Dt.W2[256:] + b_lt[t]*sum(W2[256:]) + b2)
__global__ __launch_bounds__(256) void loc_kernel(
    const _Float16* feaT,               // [B][IN][L] f16
    const _Float16* Wlt,                // [512][256] f16 padded
    const float* b_lt, const _Float16* hs, const float* W2, const float* b2,
    float* out2) {
  __shared__ float Dt[16][IN_SZ];               // 32 KB
  __shared__ __align__(16) _Float16 Atile[16 * L_SZ];   // 8 KB
  int b = blockIdx.x >> 5, mt = blockIdx.x & 31;
  int tid = threadIdx.x;
  // async-stage A tile (16 rows x 256 halfs) into LDS: 2 rounds of 4KB
  {
    const _Float16* src = Wlt + (size_t)mt * 16 * L_SZ;
#pragma unroll
    for (int rnd = 0; rnd < 2; ++rnd) {
      int off = (rnd * 256 + tid) * 8;          // halfs (16B per lane)
      async_ld_b128((uint32_t)(uintptr_t)&Atile[off], src + off);
    }
    wait_asynccnt0();
  }
  __syncthreads();

  int wave = tid >> 5, lane = tid & 31;
  int g = lane >> 4, mn = lane & 15;
  const _Float16* feaTB = feaT + (size_t)b * IN_SZ * L_SZ;
  const _Float16* arow = &Atile[(size_t)mn * L_SZ];
#pragma unroll
  for (int nti = 0; nti < 4; ++nti) {
    int nt = wave * 4 + nti;                    // 32 N-tiles over C=512
    int col = nt * 16 + mn;
    v8f c = {};
    for (int ks = 0; ks < 8; ++ks) {
      v16h a = load_a_frag(arow, ks * 32, g);
      v16h bb = load_b_frag_nk(feaTB, col, L_SZ, ks * 32, g);
      c = wmma_f16(a, bb, c);
    }
#pragma unroll
    for (int r = 0; r < 8; ++r) Dt[r + (g << 3)][col] = c[r];
  }
  __syncthreads();
  if (tid < 64) {
    int row = tid >> 2, j = tid & 3;
    int t = mt * 16 + row;
    if (t < T_STEPS) {
      const float* wrow = W2 + (size_t)j * (H_SZ + IN_SZ);
      float s = 0.f, sw = 0.f;
      for (int cc = 0; cc < IN_SZ; ++cc) {
        float w = wrow[H_SZ + cc];
        s += Dt[row][cc] * w;
        sw += w;
      }
      const _Float16* hrow = hs + ((size_t)b * T_STEPS + t) * H_SZ;
      float hd = 0.f;
      for (int hh = 0; hh < H_SZ; ++hh) hd += (float)hrow[hh] * wrow[hh];
      float val = s + b_lt[t] * sw + hd + b2[j];
      out2[((size_t)b * T_STEPS + t) * 4 + j] = fast_sigmoid(val);
    }
  }
}

// ---------------------------------------------------------------------------
extern "C" void kernel_launch(void* const* d_in, const int* in_sizes, int n_in,
                              void* d_out, int out_size, void* d_ws,
                              size_t ws_size, hipStream_t stream) {
  (void)in_sizes; (void)n_in; (void)out_size; (void)ws_size;
  const float* fea      = (const float*)d_in[0];
  const float* W_i2h    = (const float*)d_in[1];
  const float* W_h2h    = (const float*)d_in[2];
  const float* b_h2h    = (const float*)d_in[3];
  const float* w_score  = (const float*)d_in[4];
  const float* W_ih     = (const float*)d_in[5];
  const float* W_hh     = (const float*)d_in[6];
  const float* b_ih     = (const float*)d_in[7];
  const float* b_hh     = (const float*)d_in[8];
  const float* W_gen    = (const float*)d_in[9];
  const float* b_gen    = (const float*)d_in[10];
  const float* W_lt     = (const float*)d_in[11];
  const float* b_lt     = (const float*)d_in[12];
  const float* W_locgen = (const float*)d_in[13];
  const float* b_locgen = (const float*)d_in[14];

  char* base = (char*)d_ws;
  size_t off = 0;
  auto alloc = [&](size_t bytes) -> void* {
    void* p = base + off;
    off = (off + bytes + 255) & ~(size_t)255;
    return p;
  };
  const size_t nFea = (size_t)B_SZ * L_SZ * IN_SZ;     // 16.78M
  const size_t nBhp = (size_t)B_SZ * L_SZ * H_SZ;      // 8.39M
  const size_t nHs  = (size_t)B_SZ * T_STEPS * H_SZ;   // 16.42M
  _Float16* fea_h  = (_Float16*)alloc(nFea * 2);
  _Float16* feaT_h = (_Float16*)alloc(nFea * 2);
  _Float16* bHp_h  = (_Float16*)alloc(nBhp * 2);
  _Float16* hs_h   = (_Float16*)alloc(nHs * 2);
  _Float16* wi2h_h = (_Float16*)alloc((size_t)H_SZ * IN_SZ * 2);
  _Float16* wh2h_h = (_Float16*)alloc((size_t)H_SZ * H_SZ * 2);
  _Float16* wih_h  = (_Float16*)alloc((size_t)3 * H_SZ * XK * 2);
  _Float16* whh_h  = (_Float16*)alloc((size_t)3 * H_SZ * H_SZ * 2);
  _Float16* wgen_h = (_Float16*)alloc((size_t)32 * H_SZ * 2);
  _Float16* wlt_h  = (_Float16*)alloc((size_t)512 * L_SZ * 2);

  // 1) conversions + transpose
  cvt_f32_f16<<<16384, 256, 0, stream>>>(fea_h, fea, (int)nFea);
  transpose_fea<<<B_SZ * 8 * 16, 256, 0, stream>>>(fea, feaT_h);
  cvt_pad<<<(H_SZ * IN_SZ + 255) / 256, 256, 0, stream>>>(
      wi2h_h, W_i2h, H_SZ, IN_SZ, H_SZ, IN_SZ);
  cvt_pad<<<(H_SZ * H_SZ + 255) / 256, 256, 0, stream>>>(
      wh2h_h, W_h2h, H_SZ, H_SZ, H_SZ, H_SZ);
  cvt_pad<<<(3 * H_SZ * XK + 255) / 256, 256, 0, stream>>>(
      wih_h, W_ih, 3 * H_SZ, XK, 3 * H_SZ, IN_SZ + ELEM_N);
  cvt_pad<<<(3 * H_SZ * H_SZ + 255) / 256, 256, 0, stream>>>(
      whh_h, W_hh, 3 * H_SZ, H_SZ, 3 * H_SZ, H_SZ);
  cvt_pad<<<(32 * H_SZ + 255) / 256, 256, 0, stream>>>(
      wgen_h, W_gen, 32, H_SZ, ELEM_N, H_SZ);
  cvt_pad<<<(512 * L_SZ + 255) / 256, 256, 0, stream>>>(
      wlt_h, W_lt, 512, L_SZ, T_STEPS, L_SZ);

  // 2) batch_H_proj GEMM (WMMA)
  bhp_gemm<<<4096, 256, 0, stream>>>(fea_h, wi2h_h, bHp_h);

  // 3) sequential decode: 32 persistent blocks x 4 batch rows
  decode_steps<<<B_SZ / R_PB, 256, 0, stream>>>(
      fea_h, bHp_h, wh2h_h, b_h2h, w_score, wih_h, whh_h, b_ih, b_hh, wgen_h,
      b_gen, hs_h);

  // 4) structure_probs -> out[0 .. 64128*30)
  float* out_struct = (float*)d_out;
  struct_probs_kernel<<<(B_SZ * T_STEPS / 16) / 4, 256, 0, stream>>>(
      hs_h, wgen_h, b_gen, out_struct);

  // 5) loc branch -> out[64128*30 .. +64128*4)
  float* out_loc = out_struct + (size_t)B_SZ * T_STEPS * ELEM_N;
  loc_kernel<<<B_SZ * 32, 256, 0, stream>>>(feaT_h, wlt_h, b_lt, hs_h,
                                            W_locgen, b_locgen, out_loc);
}
